// PointNet2_30348238913930
// MI455X (gfx1250) — matrix operand order repaired and checked by reading, not compiled, not run on hardware
//
#include <hip/hip_runtime.h>

typedef __attribute__((ext_vector_type(16))) _Float16 v16h;
typedef __attribute__((ext_vector_type(8)))  _Float16 v8h;
typedef __attribute__((ext_vector_type(8)))  float    v8f;

#define WPB 4  // waves per block in GEMM

static inline int divUp(int a, int b) { return (a + b - 1) / b; }
static inline int ceil32(int a) { return (a + 31) & ~31; }

// ---------------- strided subsample of xyz (the ::2,::2 pooling chains) ------
__global__ void subsample_xyz_kernel(const float* __restrict__ src, float* __restrict__ dst,
                                     int B, int Hs, int Ws, int stride, int Hd, int Wd) {
  int tid = blockIdx.x * blockDim.x + threadIdx.x;
  int total = B * 3 * Hd * Wd;
  if (tid >= total) return;
  int w = tid % Wd; int t = tid / Wd;
  int h = t % Hd;   t /= Hd;
  int c = t % 3;    int b = t / 3;
  dst[tid] = src[((size_t)(b * 3 + c) * Hs + h * stride) * Ws + w * stride];
}

// ---------------- ball query: first ns indices (ascending) with d2 < r2 ------
__global__ void ball_query_kernel(const float* __restrict__ xyz,    // [B][3][N]
                                  const float* __restrict__ newxyz, // [B][3][M]
                                  int* __restrict__ idx,            // [B][M][ns]
                                  int B, int N, int M, float r2, int ns) {
  int tid = blockIdx.x * blockDim.x + threadIdx.x;
  if (tid >= B * M) return;
  int b = tid / M, m = tid % M;
  const float* xb = xyz + (size_t)b * 3 * N;
  float qx = newxyz[((size_t)b * 3 + 0) * M + m];
  float qy = newxyz[((size_t)b * 3 + 1) * M + m];
  float qz = newxyz[((size_t)b * 3 + 2) * M + m];
  int buf[32];
  int cnt = 0;
  for (int n = 0; n < N && cnt < ns; ++n) {
    float dx = xb[n] - qx;
    float dy = xb[N + n] - qy;
    float dz = xb[2 * N + n] - qz;
    float d2 = dx * dx + dy * dy + dz * dz;
    if (d2 < r2) buf[cnt++] = n;
  }
  int first = (cnt > 0) ? buf[0] : (N - 1);
  for (int j = cnt; j < ns; ++j) buf[j] = first;
  int* o = idx + (size_t)(b * M + m) * ns;
  for (int j = 0; j < ns; ++j) o[j] = buf[j];
}

// ---------------- weight prep: f32 [Co][K] -> f16 [Cop][Kp] zero-padded ------
__global__ void prep_weights_kernel(const float* __restrict__ W, const float* __restrict__ bias,
                                    _Float16* __restrict__ W16, float* __restrict__ biasP,
                                    int Co, int K, int Cop, int Kp) {
  int tid = blockIdx.x * blockDim.x + threadIdx.x;
  int total = Cop * Kp;
  if (tid < total) {
    int r = tid / Kp, k = tid % Kp;
    W16[tid] = (r < Co && k < K) ? (_Float16)W[(size_t)r * K + k] : (_Float16)0.0f;
  }
  if (tid < Cop) biasP[tid] = (tid < Co) ? bias[tid] : 0.0f;
}

// ------- gather rel-xyz + feats into f16 matrix X[S][Kp] (sample-major) ------
__global__ void gather_kernel(const float* __restrict__ xyz,    // [B][3][N]
                              const float* __restrict__ newxyz, // [B][3][M]
                              const float* __restrict__ feats,  // [B][Cf][N]
                              const int* __restrict__ idx,      // [B][M][ns]
                              _Float16* __restrict__ X,         // [S][Kp]
                              int b, int N, int M, int ns, int Cf, int Kp) {
  int Cin = Cf + 3;
  long long tid = (long long)blockIdx.x * blockDim.x + threadIdx.x;
  long long total = (long long)M * ns * Kp;
  if (tid >= total) return;
  int c = (int)(tid % Kp);
  int s = (int)(tid / Kp);
  float v = 0.0f;
  if (c < Cin) {
    int m = s / ns;
    int id = idx[(size_t)(b * M + m) * ns + (s % ns)];
    if (c < 3) v = xyz[((size_t)b * 3 + c) * N + id] - newxyz[((size_t)b * 3 + c) * M + m];
    else       v = feats[((size_t)b * Cf + (c - 3)) * N + id];
  }
  X[tid] = (_Float16)v;
}

// ---------------- GEMM: Y = relu(W16[Cop][Kp] * X[S][Kp]^T + bias) -----------
// X is sample-major, so a B fragment is 16 contiguous halves; A fragment is two
// contiguous 8-half chunks. All loads/stores unconditional b128.
// Output stored sample-major [S][Cop]; f32 path for the last sub-layer.
__global__ __launch_bounds__(32 * WPB) void gemm_relu_wmma_kernel(
    const _Float16* __restrict__ W16, const float* __restrict__ biasP,
    const _Float16* __restrict__ X, _Float16* __restrict__ Y16,
    float* __restrict__ Yf32, int Cop, int Kp, int S) {
  int lane  = threadIdx.x;                               // 0..31
  int sTile = (blockIdx.x * WPB + threadIdx.y) * 16;     // S tile base
  int cTile = blockIdx.y * 16;                           // Co tile base
  int row   = cTile + (lane & 15);                       // A row (< Cop always)
  int col   = sTile + (lane & 15);                       // B col (< S always)
  int kloA  = (lane >> 4) * 16;                          // A K sub-base per half-wave
  int kloB  = (lane >> 4) * 16;                          // B K sub-base per half-wave

  const _Float16* wrow = W16 + (size_t)row * Kp + kloA;
  const _Float16* xcol = X   + (size_t)col * Kp + kloB;

  v8f acc = {};
  for (int k0 = 0; k0 < Kp; k0 += 32) {
    v8h a0 = *(const v8h*)(wrow + k0);        // K = k0+kloA .. +7
    v8h a1 = *(const v8h*)(wrow + k0 + 8);    // K = k0+kloA+8 .. +15
    v16h av = __builtin_shufflevector(a0, a1, 0, 1, 2, 3, 4, 5, 6, 7,
                                      8, 9, 10, 11, 12, 13, 14, 15);
    v16h bv = *(const v16h*)(xcol + k0);      // K = k0+kloB .. +15 contiguous
    acc = __builtin_amdgcn_wmma_f32_16x16x32_f16(false, av, false, bv,
                                                 (short)0, acc, false, false);
  }

  // C/D layout: vgpr r -> M = cTile + r + 8*(lane>=16), N = col
  int rr = (lane >> 4) << 3;
  int ro = cTile + rr;  // 8 contiguous output rows start here
  const float4* bp = (const float4*)(biasP + ro);
  float4 b0 = bp[0], b1 = bp[1];
  float v[8];
  v[0] = acc[0] + b0.x; v[1] = acc[1] + b0.y; v[2] = acc[2] + b0.z; v[3] = acc[3] + b0.w;
  v[4] = acc[4] + b1.x; v[5] = acc[5] + b1.y; v[6] = acc[6] + b1.z; v[7] = acc[7] + b1.w;
#pragma unroll
  for (int r = 0; r < 8; ++r) v[r] = v[r] > 0.0f ? v[r] : 0.0f;

  if (Yf32) {
    float4* op = (float4*)(Yf32 + (size_t)col * Cop + ro);
    op[0] = make_float4(v[0], v[1], v[2], v[3]);
    op[1] = make_float4(v[4], v[5], v[6], v[7]);
  } else {
    v8h yv;
#pragma unroll
    for (int r = 0; r < 8; ++r) yv[r] = (_Float16)v[r];
    *(v8h*)(Y16 + (size_t)col * Cop + ro) = yv;
  }
}

// ---------------- max over nsample, scatter to output feature map ------------
__global__ void maxreduce_kernel(const float* __restrict__ Act, // [M*ns][Cop]
                                 float* __restrict__ outp,      // out + off(b,branch)
                                 int Co, int Cop, int M, int ns) {
  int tid = blockIdx.x * blockDim.x + threadIdx.x;
  if (tid >= Co * M) return;
  int c = tid % Co, m = tid / Co;
  const float* p = Act + (size_t)m * ns * Cop + c;
  float v = p[0];
  for (int j = 1; j < ns; ++j) { float t = p[(size_t)j * Cop]; v = t > v ? t : v; }
  outp[(size_t)c * M + m] = v;
}

extern "C" void kernel_launch(void* const* d_in, const int* in_sizes, int n_in,
                              void* d_out, int out_size, void* d_ws, size_t ws_size,
                              hipStream_t stream) {
  (void)in_sizes; (void)n_in; (void)out_size; (void)ws_size;
  const float* xyz0   = (const float*)d_in[0];   // (8,3,128,128)
  const float* feats0 = (const float*)d_in[1];   // (8,6,128,128)
  float* out = (float*)d_out;

  const int B = 8;
  static const int   MLP[4][2][4] = {
      {{9, 16, 16, 32},    {9, 32, 32, 64}},
      {{99, 64, 64, 128},  {99, 64, 96, 128}},
      {{259, 128, 196, 256}, {259, 128, 196, 256}},
      {{515, 256, 256, 512}, {515, 256, 384, 512}}};
  static const float RAD[4][2] = {{0.05f, 0.1f}, {0.1f, 0.2f}, {0.2f, 0.4f}, {0.4f, 0.8f}};
  static const int   NS[4][2]  = {{16, 32}, {16, 32}, {16, 32}, {16, 32}};
  const int    Ns[4]     = {16384, 1024, 256, 256};
  const int    Ms[4]     = {1024, 256, 256, 256};
  const int    Cf[4]     = {6, 96, 256, 512};
  const int    Ctot[4]   = {96, 256, 512, 1024};
  const size_t outOff[4] = {786432, 1572864, 2097152, 3145728};
  const int    cOff[4][2] = {{0, 32}, {0, 128}, {0, 256}, {0, 512}};

  // ---- workspace carve ----
  char* ws = (char*)d_ws;
  size_t off = 0;
  auto carve = [&](size_t bytes) -> void* {
    void* p = ws + off;
    off += (bytes + 255) & ~(size_t)255;
    return p;
  };
  float* xyz1 = (float*)carve((size_t)B * 3 * 1024 * 4);          // 32x32 level
  float* xyz2 = (float*)carve((size_t)B * 3 * 256 * 4);           // 16x16 level
  int* idxBuf[4][2];
  for (int l = 0; l < 4; ++l)
    for (int br = 0; br < 2; ++br)
      idxBuf[l][br] = (int*)carve((size_t)B * Ms[l] * NS[l][br] * 4);
  _Float16* wBuf[3];
  float*    bBuf[3];
  for (int s = 0; s < 3; ++s) {
    wBuf[s] = (_Float16*)carve((size_t)512 * 544 * 2);            // max Cop*Kp
    bBuf[s] = (float*)carve((size_t)512 * 4);
  }
  _Float16* Xbuf = (_Float16*)carve((size_t)8192 * 544 * 2);      // max S*Kp
  _Float16* actA = (_Float16*)carve((size_t)8192 * 512 * 2);      // max S*Cop
  _Float16* actB = (_Float16*)carve((size_t)8192 * 512 * 2);
  float*    actF = (float*)carve((size_t)8192 * 512 * 4);         // last stage f32

  // ---- l_feat[0] = input features ----
  hipMemcpyAsync(out, feats0, (size_t)786432 * 4, hipMemcpyDeviceToDevice, stream);

  // ---- xyz pyramid ----
  subsample_xyz_kernel<<<divUp(B * 3 * 32 * 32, 256), 256, 0, stream>>>(
      xyz0, xyz1, B, 128, 128, 4, 32, 32);
  subsample_xyz_kernel<<<divUp(B * 3 * 16 * 16, 256), 256, 0, stream>>>(
      xyz0, xyz2, B, 128, 128, 8, 16, 16);

  const float* xyzP[4]  = {xyz0, xyz1, xyz2, xyz2};
  const float* newP[4]  = {xyz1, xyz2, xyz2, xyz2};
  const float* featP[4] = {feats0, out + outOff[0], out + outOff[1], out + outOff[2]};

  // ---- ball queries (all batches per launch) ----
  for (int l = 0; l < 4; ++l)
    for (int br = 0; br < 2; ++br) {
      float r = RAD[l][br];
      ball_query_kernel<<<divUp(B * Ms[l], 128), 128, 0, stream>>>(
          xyzP[l], newP[l], idxBuf[l][br], B, Ns[l], Ms[l], r * r, NS[l][br]);
    }

  auto Wp = [&](int l, int br, int s) -> const float* {
    return (const float*)d_in[2 + ((l * 2 + br) * 3 + s) * 2];
  };
  auto Bp = [&](int l, int br, int s) -> const float* {
    return (const float*)d_in[2 + ((l * 2 + br) * 3 + s) * 2 + 1];
  };

  for (int l = 0; l < 4; ++l) {
    for (int br = 0; br < 2; ++br) {
      int ns  = NS[l][br];
      int S   = Ms[l] * ns;
      int Kp0 = ceil32(MLP[l][br][0]);
      int Cop[3], Kp[3];
      for (int s = 0; s < 3; ++s) {
        Kp[s]  = (s == 0) ? Kp0 : Cop[s - 1];
        Cop[s] = ceil32(MLP[l][br][s + 1]);
        // pack weights once per branch (amortized over batches)
        prep_weights_kernel<<<divUp(Cop[s] * Kp[s], 256), 256, 0, stream>>>(
            Wp(l, br, s), Bp(l, br, s), wBuf[s], bBuf[s],
            MLP[l][br][s + 1], MLP[l][br][s], Cop[s], Kp[s]);
      }
      for (int b = 0; b < B; ++b) {
        // gather (also writes K-padding zeros)
        long long tot = (long long)S * Kp0;
        gather_kernel<<<(int)((tot + 255) / 256), 256, 0, stream>>>(
            xyzP[l], newP[l], featP[l], idxBuf[l][br], Xbuf,
            b, Ns[l], Ms[l], ns, Cf[l], Kp0);
        // three GEMM+ReLU stages, all loads/stores unconditional
        const _Float16* in16 = Xbuf;
        for (int s = 0; s < 3; ++s) {
          bool last = (s == 2);
          _Float16* o16 = (s == 0) ? actA : actB;
          dim3 grid(S / (16 * WPB), Cop[s] / 16);
          dim3 block(32, WPB);
          gemm_relu_wmma_kernel<<<grid, block, 0, stream>>>(
              wBuf[s], bBuf[s], in16,
              last ? nullptr : o16, last ? actF : nullptr,
              Cop[s], Kp[s], S);
          in16 = o16;
        }
        // max over nsample -> d_out
        int Co = MLP[l][br][3];
        float* outp = out + outOff[l] + (size_t)(b * Ctot[l] + cOff[l][br]) * Ms[l];
        maxreduce_kernel<<<divUp(Co * Ms[l], 256), 256, 0, stream>>>(
            actF, outp, Co, Cop[2], Ms[l], ns);
      }
    }
  }
}
